// ModulatedConv2d_45028437131960
// MI455X (gfx1250) — matrix operand order repaired and checked by reading, not compiled, not run on hardware
//
#include <hip/hip_runtime.h>

typedef __attribute__((ext_vector_type(16))) __bf16       v16bf;
typedef __attribute__((ext_vector_type(8)))  float        v8f;
typedef __attribute__((ext_vector_type(8)))  unsigned int v8u;
typedef int v4i __attribute__((vector_size(16)));

#define C_IN   512
#define O_CH   512
#define IMH    64
#define IMW    64
#define MODSC  0.04419417382415922f     // 1/sqrt(512)
#define CONVSC 0.014731391274719742f    // 1/sqrt(512*9)
#define EPSV   1e-8f

#define AS1 __attribute__((address_space(1)))
#define AS3 __attribute__((address_space(3)))

#if __has_builtin(__builtin_amdgcn_global_load_async_to_lds_b128)
#define HAVE_ASYNC 1
#else
#define HAVE_ASYNC 0
#endif

union FragU { v8u u; v16bf b; };

__device__ __forceinline__ unsigned int pack_bf16x2(float a, float b) {
#if __has_builtin(__builtin_amdgcn_cvt_pk_bf16_f32)
  auto r = __builtin_amdgcn_cvt_pk_bf16_f32(a, b);
  return __builtin_bit_cast(unsigned int, r);
#else
  unsigned int ua = __builtin_bit_cast(unsigned int, a);
  unsigned int ub = __builtin_bit_cast(unsigned int, b);
  ua += 0x7FFFu + ((ua >> 16) & 1u);    // round-to-nearest-even
  ub += 0x7FFFu + ((ub >> 16) & 1u);
  return (ua >> 16) | (ub & 0xFFFF0000u);
#endif
}

__device__ __forceinline__ void async_cp16(const unsigned int* g, unsigned int* l) {
#if HAVE_ASYNC
  __builtin_amdgcn_global_load_async_to_lds_b128(
      (AS1 v4i*)g, (AS3 v4i*)l, 0, 0);
#else
  *(uint4*)l = *(const uint4*)g;
#endif
}

__device__ __forceinline__ void async_wait() {
#if HAVE_ASYNC
#if __has_builtin(__builtin_amdgcn_s_wait_asynccnt)
  __builtin_amdgcn_s_wait_asynccnt(0);
#else
  asm volatile("s_wait_asynccnt 0x0" ::: "memory");
#endif
#endif
}

// s[b][c] = MODSC * <style[b,:], mod_weight[c,:]> + mod_bias[c]
__global__ __launch_bounds__(512) void modulate_kernel(
    const float* __restrict__ style, const float* __restrict__ mw,
    const float* __restrict__ mb, float* __restrict__ s_out) {
  __shared__ float st[512];
  const int b = blockIdx.x, c = threadIdx.x;
  st[c] = style[b * 512 + c];
  __syncthreads();
  const float* row = mw + (size_t)c * 512;
  float acc = 0.f;
#pragma unroll 8
  for (int d = 0; d < 512; ++d) acc += st[d] * row[d];
  s_out[b * 512 + c] = acc * MODSC + mb[c];
}

// wb[p][cp][o] = bf16x2(CONVSC*w[o][2cp][p], CONVSC*w[o][2cp+1][p])  (GEMM-B layout)
// w2t[c][o]    = sum_p (CONVSC*w[o][c][p])^2
__global__ __launch_bounds__(256) void prep_weight_kernel(
    const float* __restrict__ wgt, unsigned int* __restrict__ wb,
    float* __restrict__ w2t) {
  const int t  = blockIdx.x * 256 + threadIdx.x;   // 0..131071
  const int o  = t & 511;
  const int cp = t >> 9;                           // 0..255
  const int c0 = cp * 2;
  const float* w0p = wgt + ((size_t)o * 512 + c0) * 9;
  const float* w1p = w0p + 9;
  float s0 = 0.f, s1 = 0.f;
#pragma unroll
  for (int p = 0; p < 9; ++p) {
    float a = w0p[p] * CONVSC;
    float q = w1p[p] * CONVSC;
    s0 += a * a; s1 += q * q;
    wb[((size_t)p * 256 + cp) * 512 + o] = pack_bf16x2(a, q);
  }
  w2t[(size_t)c0 * 512 + o]       = s0;
  w2t[(size_t)(c0 + 1) * 512 + o] = s1;
}

// demod[b][o] = rsqrt(eps + sum_c s[b][c]^2 * w2t[c][o])
__global__ __launch_bounds__(512) void demod_kernel(
    const float* __restrict__ s, const float* __restrict__ w2t,
    float* __restrict__ dmd) {
  __shared__ float s2[512];
  const int b = blockIdx.x, o = threadIdx.x;
  float sv = s[b * 512 + o];
  s2[o] = sv * sv;
  __syncthreads();
  float acc = EPSV;
#pragma unroll 8
  for (int c = 0; c < 512; ++c) acc += s2[c] * w2t[(size_t)c * 512 + o];
  dmd[b * 512 + o] = rsqrtf(acc);
}

// Pre-pass: xmb[b][cpair][h][w] = bf16x2(x[b][2cp][h][w]*s[b][2cp], x[b][2cp+1][h][w]*s[b][2cp+1])
__global__ __launch_bounds__(256) void xmb_pack_kernel(
    const float* __restrict__ x, const float* __restrict__ s,
    unsigned int* __restrict__ xmb) {
  const size_t t = (size_t)blockIdx.x * 256 + threadIdx.x;  // 16*256*4096 total
  const int hw = (int)(t & 4095);
  const int bc = (int)(t >> 12);
  const int cp = bc & 255;
  const int b  = bc >> 8;
  const int c0 = cp * 2;
  const float s0 = s[b * 512 + c0];
  const float s1 = s[b * 512 + c0 + 1];
  const float a0 = x[((size_t)(b * 512 + c0)) * 4096 + hw] * s0;
  const float a1 = x[((size_t)(b * 512 + c0 + 1)) * 4096 + hw] * s1;
  xmb[t] = pack_bf16x2(a0, a1);
}

// Implicit-GEMM modulated conv: block tile = 128 pixels (2 rows) x 64 out-ch.
// 4 waves; wave tile 64(M) x 32(N) = 4x2 fragments of 16x16 f32 via v_wmma_f32_16x16x32_bf16.
template <bool USE_XMB>
__global__ __launch_bounds__(128) void modconv_wmma_kernel(
    const float* __restrict__ x, const unsigned int* __restrict__ xmb,
    const float* __restrict__ smod_g, const float* __restrict__ dmd,
    const unsigned int* __restrict__ wb, float* __restrict__ out) {
  __shared__ unsigned int XS[4 * 16 * 72];   // [r(4)][cpair(16)][72]; interior u32 idx 4..67, halo 3 & 68
  __shared__ unsigned int WS[9 * 16 * 68];   // [p(9)][cpair(16)][o(68 pad)] bf16x2
  __shared__ float SMOD[512];

  const int o_base = blockIdx.x * 64;
  const int h0     = blockIdx.y * 2;
  const int b      = blockIdx.z;
  const int tid  = threadIdx.x;
  const int lane = tid & 31;
  const int wave = tid >> 5;
  const int l16  = lane & 15;
  const int lhi  = lane >> 4;
  const int wm   = wave >> 1;     // M-wave: 0..1 (offset wm*64)
  const int wn   = wave & 1;      // N-wave: 0..1 (offset wn*32)

  if constexpr (!USE_XMB) {
    for (int i = tid; i < 512; i += 128) SMOD[i] = smod_g[b * 512 + i];
  }
  // halo columns (w_in = -1 and 64) are always zero
  for (int i = tid; i < 64; i += 128) { XS[i * 72 + 3] = 0u; XS[i * 72 + 68] = 0u; }
  // rows outside the image are always zero; staged rows never touch them
#pragma unroll
  for (int r = 0; r < 4; ++r) {
    const int hh = h0 - 1 + r;
    if (hh < 0 || hh >= IMH) {
      for (int i = tid; i < 16 * 72; i += 128) XS[r * 16 * 72 + i] = 0u;
    }
  }

  v8f acc[4][2];
  {
    v8f z = {0.f, 0.f, 0.f, 0.f, 0.f, 0.f, 0.f, 0.f};
#pragma unroll
    for (int i = 0; i < 4; ++i)
#pragma unroll
      for (int j = 0; j < 2; ++j) acc[i][j] = z;
  }

  const float* xb = x + (size_t)b * C_IN * IMH * IMW;

  for (int cc = 0; cc < C_IN; cc += 32) {
    __syncthreads();   // previous chunk's fragment reads done before restage

    // --- stage modulated-x patch: 4 rows x 16 cpairs x 64 cols (bf16x2) ---
    if constexpr (USE_XMB) {
      // pure byte-moving: async b128 copies from the packed pre-pass buffer
#pragma unroll
      for (int i = 0; i < 8; ++i) {
        const int e   = tid + i * 128;   // 0..1023
        const int q4  = e & 15;          // 16 x b128 per (r,cp) row
        const int row = e >> 4;          // 0..63 = r*16+cp
        const int r   = row >> 4;
        const int cp  = row & 15;
        const int hh  = h0 - 1 + r;
        if (hh >= 0 && hh < IMH) {
          const unsigned int* g = xmb +
              (((size_t)(b * 256 + (cc >> 1) + cp) * IMH + hh) * IMW) + q4 * 4;
          async_cp16(g, &XS[row * 72 + 4 + q4 * 4]);
        }
      }
    } else {
#pragma unroll 4
      for (int i = 0; i < 32; ++i) {
        const int e   = tid + i * 128;
        const int w   = e & 63;
        const int rcp = e >> 6;          // 0..63
        const int r   = rcp >> 4;
        const int cp  = rcp & 15;
        const int hh  = h0 - 1 + r;
        const int c0  = cc + cp * 2;
        if (hh >= 0 && hh < IMH) {
          const float* rp = xb + ((size_t)c0 * IMH + hh) * IMW;
          const float a0 = rp[w] * SMOD[c0];
          const float a1 = rp[IMH * IMW + w] * SMOD[c0 + 1];
          XS[rcp * 72 + w + 4] = pack_bf16x2(a0, a1);
        }
      }
    }
    // --- stage weight slab: 9 pos x 16 cpairs x 64 out-ch (async b128) ---
#pragma unroll 3
    for (int i = 0; i < 18; ++i) {
      const int e4  = tid + i * 128;   // 0..2303
      const int o4  = e4 & 15;
      const int row = e4 >> 4;         // 0..143 = p*16 + cpl
      const int p   = row >> 4;
      const int cpl = row & 15;
      const unsigned int* g = wb +
          ((size_t)(p * 256 + (cc >> 1) + cpl)) * 512 + o_base + o4 * 4;
      async_cp16(g, &WS[row * 68 + o4 * 4]);
    }
    async_wait();
    __syncthreads();

    // --- 9 kernel taps, each a K=32 bf16 WMMA chain ---
#pragma unroll
    for (int p = 0; p < 9; ++p) {
      const int pr = p / 3, ps = p % 3;
      // B fragments (32x16): lane<16 holds K=2v,2v+1 ; lane>=16 holds K=16+2v,..
      FragU bfr[2];
#pragma unroll
      for (int j = 0; j < 2; ++j) {
        const int col  = wn * 32 + j * 16 + l16;
        const int rowb = p * 16 + lhi * 8;
#pragma unroll
        for (int v = 0; v < 8; ++v) bfr[j].u[v] = WS[(rowb + v) * 68 + col];
      }
      // A fragments (16x32): K-order {0..15} / {16..31}
      FragU afr[4];
#pragma unroll
      for (int i2 = 0; i2 < 4; ++i2) {
        const int m    = wm * 64 + i2 * 16 + l16;
        const int r    = (m >> 6) + pr;
        const int widx = (m & 63) + ps + 3;
        const int cpb  = lhi * 8;
#pragma unroll
        for (int v = 0; v < 8; ++v) {
          const int cp = v + cpb;
          afr[i2].u[v] = XS[(r * 16 + cp) * 72 + widx];
        }
      }
#pragma unroll
      for (int i2 = 0; i2 < 4; ++i2)
#pragma unroll
        for (int j = 0; j < 2; ++j)
          acc[i2][j] = __builtin_amdgcn_wmma_f32_16x16x32_bf16(
              false, afr[i2].b, false, bfr[j].b,
              (short)0, acc[i2][j], false, false);
    }
  }

  // --- epilogue: demod scale + store (C/D layout: VGPR q -> M=q+8*lhi, N=l16) ---
#pragma unroll
  for (int i2 = 0; i2 < 4; ++i2) {
#pragma unroll
    for (int j = 0; j < 2; ++j) {
      const int o = o_base + wn * 32 + j * 16 + l16;
      const float dm = dmd[b * O_CH + o];
#pragma unroll
      for (int q = 0; q < 8; ++q) {
        const int m = wm * 64 + i2 * 16 + q + lhi * 8;
        const int h = h0 + (m >> 6);
        const int w = m & 63;
        out[(((size_t)b * O_CH + o) * IMH + h) * IMW + w] = acc[i2][j][q] * dm;
      }
    }
  }
}

extern "C" void kernel_launch(void* const* d_in, const int* in_sizes, int n_in,
                              void* d_out, int out_size, void* d_ws, size_t ws_size,
                              hipStream_t stream) {
  const float* x          = (const float*)d_in[0];   // (16,512,64,64)
  const float* style      = (const float*)d_in[1];   // (16,512)
  const float* weight     = (const float*)d_in[2];   // (1,512,512,3,3)
  const float* mod_weight = (const float*)d_in[3];   // (512,512)
  const float* mod_bias   = (const float*)d_in[4];   // (512,)
  float* out = (float*)d_out;

  char* ws = (char*)d_ws;
  float*        s_mod = (float*)ws;                              //  32 KB
  float*        dmd   = (float*)(ws + 32768);                    //  32 KB
  float*        w2t   = (float*)(ws + 65536);                    //   1 MB
  unsigned int* wb    = (unsigned int*)(ws + 65536 + 1048576);   // 4.5 MB
  unsigned int* xmb   = (unsigned int*)(ws + 5832704);           //  64 MB (optional)
  const bool use_xmb = ws_size >= 72941568ull;                   // base + xmb

  modulate_kernel<<<16, 512, 0, stream>>>(style, mod_weight, mod_bias, s_mod);
  prep_weight_kernel<<<512, 256, 0, stream>>>(weight, wb, w2t);
  demod_kernel<<<16, 512, 0, stream>>>(s_mod, w2t, dmd);

  dim3 grid(8 /*o tiles*/, 32 /*row pairs*/, 16 /*batch*/);
  if (use_xmb) {
    xmb_pack_kernel<<<65536, 256, 0, stream>>>(x, s_mod, xmb);
    modconv_wmma_kernel<true><<<grid, 128, 0, stream>>>(x, xmb, s_mod, dmd, wb, out);
  } else {
    modconv_wmma_kernel<false><<<grid, 128, 0, stream>>>(x, xmb, s_mod, dmd, wb, out);
  }
}
